// BertSelfOutput_31963146617405
// MI455X (gfx1250) — 8-bit matrix operand order repaired and checked by reading, compiled, not run on hardware
//
#include <hip/hip_runtime.h>

typedef int v8i __attribute__((ext_vector_type(8)));

#define H       1024
#define LDS_PAD 16
#define AROWB   (H + LDS_PAD)
#define CLIPV   2.5f
#define QMAX    127.0f
#define LN_EPS  1e-12f

__device__ __forceinline__ float clip_abs(float v) {
    return fminf(fabsf(v), CLIPV);
}

__device__ __forceinline__ int quant1(float v, float s) {
    return (int)rintf(fminf(fmaxf(v, -CLIPV), CLIPV) * s);
}

__global__ void absmax_kernel(const float* __restrict__ x, int n4,
                              unsigned* __restrict__ out) {
    const float4* x4 = (const float4*)x;
    float m = 0.0f;
    for (int i = blockIdx.x * blockDim.x + threadIdx.x; i < n4;
         i += gridDim.x * blockDim.x) {
        float4 v = x4[i];
        m = fmaxf(m, fmaxf(fmaxf(clip_abs(v.x), clip_abs(v.y)),
                           fmaxf(clip_abs(v.z), clip_abs(v.w))));
    }
    for (int off = 16; off > 0; off >>= 1)
        m = fmaxf(m, __shfl_xor(m, off, 32));
    __shared__ float smax[16];
    const int wave = threadIdx.x >> 5, lane = threadIdx.x & 31;
    if (lane == 0) smax[wave] = m;
    __syncthreads();
    if (threadIdx.x == 0) {
        float r = 0.0f;
        const int nw = blockDim.x >> 5;
        for (int i = 0; i < nw; ++i) r = fmaxf(r, smax[i]);
        atomicMax(out, __float_as_uint(r));
    }
}

__global__ void quantw_kernel(const float* __restrict__ w, int n4,
                              const unsigned* __restrict__ scales,
                              int* __restrict__ wq) {
    const int i = blockIdx.x * blockDim.x + threadIdx.x;
    if (i >= n4) return;
    const float s = QMAX / fmaxf(__uint_as_float(scales[1]), 1e-8f);
    const float4 v = ((const float4*)w)[i];
    const int q0 = quant1(v.x, s), q1 = quant1(v.y, s);
    const int q2 = quant1(v.z, s), q3 = quant1(v.w, s);
    wq[i] = (q0 & 0xff) | ((q1 & 0xff) << 8) | ((q2 & 0xff) << 16) |
            ((q3 & 0xff) << 24);
}

__launch_bounds__(512)
__global__ void fused_kernel(const float* __restrict__ hid,
                             const float* __restrict__ resid,
                             const signed char* __restrict__ wq,
                             const float* __restrict__ bias,
                             const float* __restrict__ gamma,
                             const float* __restrict__ beta,
                             const unsigned* __restrict__ scales,
                             float* __restrict__ out) {
    __shared__ __align__(16) unsigned char ldsA[16 * AROWB];
    __shared__ float rsum[16];
    __shared__ float rsum2[16];

    const int tid  = threadIdx.x;
    const int wave = tid >> 5;
    const int lane = tid & 31;
    const int m0   = blockIdx.x * 16;

    const float sh  = QMAX / fmaxf(__uint_as_float(scales[0]), 1e-8f);
    const float sw  = QMAX / fmaxf(__uint_as_float(scales[1]), 1e-8f);
    const float dqs = 1.0f / (sh * sw);

    {
        const float4* h4 = (const float4*)(hid + (size_t)m0 * H);
        unsigned* ldsA32 = (unsigned*)ldsA;
        for (int idx = tid; idx < 16 * H / 4; idx += 512) {
            float4 v = h4[idx];
            const int q0 = quant1(v.x, sh), q1 = quant1(v.y, sh);
            const int q2 = quant1(v.z, sh), q3 = quant1(v.w, sh);
            const int row = idx >> 8, kd = idx & 255;
            ldsA32[row * (AROWB / 4) + kd] =
                (unsigned)((q0 & 0xff) | ((q1 & 0xff) << 8) |
                           ((q2 & 0xff) << 16) | ((q3 & 0xff) << 24));
        }
    }
    if (tid < 16) { rsum[tid] = 0.0f; rsum2[tid] = 0.0f; }
    __syncthreads();

    v8i acc[4] = {};
    const int nbase = wave * 64;
    const int arow  = lane & 15;
    const int ahalf = (lane >> 4) << 4;
    const int bhalf = (lane >> 4) << 4;
    const int col0  = lane & 15;

#pragma unroll 1
    for (int kk = 0; kk < H / 64; ++kk) {
        v8i a;
        {
            const unsigned char* base = ldsA + arow * AROWB + kk * 64 + ahalf;
            const uint4 d0 = *(const uint4*)(base + 0);
            const uint4 d1 = *(const uint4*)(base + 32);
            a[0] = (int)d0.x; a[1] = (int)d0.y; a[2] = (int)d0.z; a[3] = (int)d0.w;
            a[4] = (int)d1.x; a[5] = (int)d1.y; a[6] = (int)d1.z; a[7] = (int)d1.w;
        }
        v8i bf[4];
#pragma unroll
        for (int t = 0; t < 4; ++t) {
            const int ncol = nbase + t * 16 + col0;
            const signed char* bp = wq + (size_t)ncol * H + kk * 64 + bhalf;
            const uint4 blo = *(const uint4*)(bp);
            const uint4 bhi = *(const uint4*)(bp + 32);
            bf[t][0] = (int)blo.x; bf[t][1] = (int)blo.y;
            bf[t][2] = (int)blo.z; bf[t][3] = (int)blo.w;
            bf[t][4] = (int)bhi.x; bf[t][5] = (int)bhi.y;
            bf[t][6] = (int)bhi.z; bf[t][7] = (int)bhi.w;
        }
#pragma unroll
        for (int t = 0; t < 4; ++t) {
            acc[t] = __builtin_amdgcn_wmma_i32_16x16x64_iu8(
                true, a, true, bf[t], acc[t], false, false);
        }
    }

    const int rowoff = (lane >> 4) ? 8 : 0;
    float yv[4][8];
#pragma unroll
    for (int t = 0; t < 4; ++t) {
        const int col = nbase + t * 16 + col0;
        const float bcol = bias[col];
#pragma unroll
        for (int j = 0; j < 8; ++j) {
            const int r = rowoff + j;
            yv[t][j] = (float)acc[t][j] * dqs + bcol +
                       resid[(size_t)(m0 + r) * H + col];
        }
    }
#pragma unroll
    for (int j = 0; j < 8; ++j) {
        float p  = yv[0][j] + yv[1][j] + yv[2][j] + yv[3][j];
        float p2 = yv[0][j] * yv[0][j] + yv[1][j] * yv[1][j] +
                   yv[2][j] * yv[2][j] + yv[3][j] * yv[3][j];
#pragma unroll
        for (int m = 1; m < 16; m <<= 1) {
            p  += __shfl_xor(p, m, 32);
            p2 += __shfl_xor(p2, m, 32);
        }
        if (col0 == 0) {
            atomicAdd(&rsum[rowoff + j], p);
            atomicAdd(&rsum2[rowoff + j], p2);
        }
    }
    __syncthreads();

#pragma unroll
    for (int t = 0; t < 4; ++t) {
        const int col = nbase + t * 16 + col0;
        const float g = gamma[col], bb = beta[col];
#pragma unroll
        for (int j = 0; j < 8; ++j) {
            const int r = rowoff + j;
            const float mu   = rsum[r] * (1.0f / H);
            const float var  = rsum2[r] * (1.0f / H) - mu * mu;
            const float rstd = rsqrtf(var + LN_EPS);
            out[(size_t)(m0 + r) * H + col] = (yv[t][j] - mu) * rstd * g + bb;
        }
    }
}

extern "C" void kernel_launch(void* const* d_in, const int* in_sizes, int n_in,
                              void* d_out, int out_size, void* d_ws,
                              size_t ws_size, hipStream_t stream) {
    (void)n_in; (void)out_size; (void)ws_size;
    const float* hid  = (const float*)d_in[0];
    const float* res  = (const float*)d_in[1];
    const float* wgt  = (const float*)d_in[2];
    const float* bias = (const float*)d_in[3];
    const float* gam  = (const float*)d_in[4];
    const float* bet  = (const float*)d_in[5];
    float* out = (float*)d_out;

    const int nh = in_sizes[0];
    const int M  = nh / H;

    unsigned*    scales = (unsigned*)d_ws;
    signed char* wq     = (signed char*)d_ws + 256;

    hipMemsetAsync(d_ws, 0, 256, stream);
    absmax_kernel<<<2048, 256, 0, stream>>>(hid, nh / 4, scales + 0);
    absmax_kernel<<<256, 256, 0, stream>>>(wgt, (H * H) / 4, scales + 1);
    quantw_kernel<<<(H * H / 4 + 255) / 256, 256, 0, stream>>>(
        wgt, H * H / 4, scales, (int*)wq);
    fused_kernel<<<M / 16, 512, 0, stream>>>(hid, res, wq, bias, gam, bet,
                                             scales, out);
}
